// FourierAttention_9113920602750
// MI455X (gfx1250) — compile-verified, matrix operand order checked by reading, not run on hardware
//
#include <hip/hip_runtime.h>
#include <hip/hip_bf16.h>

// FourierAttention on MI455X (gfx1250):
// sparse-mode DFT -> per-mode complex mix -> sparse-mode inverse DFT,
// all as bf16 WMMA (v_wmma_f32_16x16x32_bf16) with f32 accumulation.
//
// Sizes fixed by the reference: B=16,H=8 -> BH=128; S=4096; E=64; MODES=64.

#define S_LEN 4096
#define E_DIM 64
#define NMODE 64
#define BH    128

typedef __attribute__((ext_vector_type(16))) __bf16        v16bf;
typedef __attribute__((ext_vector_type(8)))  float         v8f;
typedef __attribute__((ext_vector_type(4)))  unsigned int  u32x4;
typedef __attribute__((ext_vector_type(4)))  float         f32x4;

union FragU { v16bf v; u32x4 q[2]; };

#define WMMA_BF16(a, b, c) \
  __builtin_amdgcn_wmma_f32_16x16x32_bf16(false, (a), false, (b), (short)0, (c), false, false)

__device__ __forceinline__ unsigned short f2bf(float f) {
  unsigned int u = __float_as_uint(f);
  u += 0x7fffu + ((u >> 16) & 1u);           // round-to-nearest-even
  return (unsigned short)(u >> 16);
}

// Load one 16x32 bf16 A/B fragment (wave32 layout):
// lane L (m = L&15, half = L>>4) holds row (m_row0+m), K chunks
// [k0+8*half, +8) and [k0+16+8*half, +8)  -> two 16-byte loads.
__device__ __forceinline__ v16bf load_frag(const unsigned short* base, int row_stride,
                                           int m_row0, int k0, int lane) {
  int m = lane & 15, half = lane >> 4;
  const unsigned short* p = base + (size_t)(m_row0 + m) * row_stride + k0 + 8 * half;
  FragU u;
  u.q[0] = *(const u32x4*)p;
  u.q[1] = *(const u32x4*)(p + 16);
  return u.v;
}

// ---------------------------------------------------------------------------
// Kernel 0: precompute bf16 twiddle matrices and transposed weights.
//   F2[128][4096]: rows 0..63 = cos(2*pi*k_i*s/S), rows 64..127 = -sin(...)
//   G [4096][128]: cols 0..63 = c_i*cos, cols 64..127 = -c_i*sin, c_i=(k?2:1)/S
//   Wre/Wim/WimN[i][o][e] = (transposed, WimN negated) weights for the mix.
// ---------------------------------------------------------------------------
__global__ void __launch_bounds__(256)
fa_precompute(const float* __restrict__ wr, const float* __restrict__ wi,
              const int* __restrict__ index,
              unsigned short* __restrict__ F2, unsigned short* __restrict__ G,
              unsigned short* __restrict__ Wre, unsigned short* __restrict__ Wim,
              unsigned short* __restrict__ WimN) {
  const float TPS = 6.2831853071795864769f / (float)S_LEN;
  int t = blockIdx.x * blockDim.x + threadIdx.x;
  const int totF = 2 * NMODE * S_LEN;     // 524288
  const int totG = S_LEN * 2 * NMODE;     // 524288
  const int totW = NMODE * E_DIM * E_DIM; // 262144
  if (t < totF) {
    int row = t >> 12, s = t & (S_LEN - 1);
    int k = index[row & (NMODE - 1)];
    float th = (float)((k * s) & (S_LEN - 1)) * TPS;  // exact phase mod S
    float v = (row < NMODE) ? __cosf(th) : -__sinf(th);
    F2[t] = f2bf(v);
    return;
  }
  t -= totF;
  if (t < totG) {
    int s = t >> 7, kc = t & 127;
    int k = index[kc & (NMODE - 1)];
    float c = ((k == 0) ? 1.0f : 2.0f) * (1.0f / (float)S_LEN);
    float th = (float)((k * s) & (S_LEN - 1)) * TPS;
    float v = (kc < NMODE) ? c * __cosf(th) : -c * __sinf(th);
    G[t] = f2bf(v);
    return;
  }
  t -= totG;
  if (t < totW) {
    int e = t & 63, o = (t >> 6) & 63, i = t >> 12;
    int src = (e * E_DIM + o) * NMODE + i;            // weights are [e][o][i]
    Wre[t] = f2bf(wr[src]);
    float im = wi[src];
    Wim[t]  = f2bf(im);
    WimN[t] = f2bf(-im);
  }
}

// ---------------------------------------------------------------------------
// Kernel 1: forward sparse DFT.  Per (bh, e-half) block:
//   C[128 rows, 32 e] += F2[128,4096] @ x_bh[4096, e0:e0+32]
// x chunk (32 k x 32 e, f32) is transposed+converted through LDS so that
// B fragments ([N=e][K=s]) are contiguous 16-byte reads.
// Writes bf16 Xre/Xim laid out [mode][bh][e] for the mix stage.
// ---------------------------------------------------------------------------
__global__ void __launch_bounds__(256)
fa_dft(const float* __restrict__ x, const unsigned short* __restrict__ F2,
       unsigned short* __restrict__ Xre, unsigned short* __restrict__ Xim) {
  __shared__ unsigned short xT[32 * 32];   // [e_local][kk], bf16
  int bh = blockIdx.x;
  int e0 = blockIdx.y * 32;
  int tid = threadIdx.x;
  int lane = tid & 31, wave = tid >> 5;    // 8 waves; wave = M-tile (16 mode-rows)
  const float* xb = x + (size_t)bh * S_LEN * E_DIM;

  v8f acc0 = {}; v8f acc1 = {};
  for (int k0 = 0; k0 < S_LEN; k0 += 32) {
    __syncthreads();
    {   // stage 32x32 f32 chunk -> bf16 transposed in LDS
      int flat = tid * 4;
      int kk = flat >> 5, ej = flat & 31;
      const float* src = xb + (size_t)(k0 + kk) * E_DIM + e0 + ej;
      f32x4 d = *(const f32x4*)src;
      xT[(ej + 0) * 32 + kk] = f2bf(d.x);
      xT[(ej + 1) * 32 + kk] = f2bf(d.y);
      xT[(ej + 2) * 32 + kk] = f2bf(d.z);
      xT[(ej + 3) * 32 + kk] = f2bf(d.w);
      if (k0 + 32 < S_LEN)
        __builtin_prefetch(src + 32 * E_DIM, 0, 1);   // global_prefetch_b8
    }
    __syncthreads();
    v16bf a  = load_frag(F2, S_LEN, wave * 16, k0, lane);
    v16bf b0 = load_frag(xT, 32, 0,  0, lane);
    v16bf b1 = load_frag(xT, 32, 16, 0, lane);
    acc0 = WMMA_BF16(a, b0, acc0);
    acc1 = WMMA_BF16(a, b1, acc1);
  }

  int m = lane & 15, half = lane >> 4;
  for (int v = 0; v < 8; v++) {
    int r = wave * 16 + v + 8 * half;      // mode-row (0..63 re, 64..127 im)
    unsigned short* dst = (r < NMODE) ? Xre : Xim;
    size_t o = ((size_t)(r & (NMODE - 1)) * BH + bh) * E_DIM + e0 + m;
    dst[o]      = f2bf(acc0[v]);
    dst[o + 16] = f2bf(acc1[v]);
  }
}

// ---------------------------------------------------------------------------
// Kernel 2: per-mode complex channel mix.  One block per mode:
//   Yre = Xre@Wre + Xim@(-Wim);  Yim = Xre@Wim + Xim@Wre   (all [128bh x 64o])
// Writes bf16 Yt[bh][o][k] with k=i (real part) / k=64+i (imag part),
// i.e. exactly the B-matrix layout the inverse stage consumes.
// ---------------------------------------------------------------------------
__global__ void __launch_bounds__(256)
fa_mix(const unsigned short* __restrict__ Xre, const unsigned short* __restrict__ Xim,
       const unsigned short* __restrict__ Wre, const unsigned short* __restrict__ Wim,
       const unsigned short* __restrict__ WimN, unsigned short* __restrict__ Yt) {
  int i = blockIdx.x;
  int lane = threadIdx.x & 31, wave = threadIdx.x >> 5;  // wave = bh tile
  const unsigned short* xre = Xre + (size_t)i * BH * E_DIM;
  const unsigned short* xim = Xim + (size_t)i * BH * E_DIM;
  const unsigned short* wre = Wre + (size_t)i * E_DIM * E_DIM;
  const unsigned short* wim = Wim + (size_t)i * E_DIM * E_DIM;
  const unsigned short* wmn = WimN + (size_t)i * E_DIM * E_DIM;
  int row0 = wave * 16;
  v16bf aRe0 = load_frag(xre, E_DIM, row0, 0,  lane);
  v16bf aRe1 = load_frag(xre, E_DIM, row0, 32, lane);
  v16bf aIm0 = load_frag(xim, E_DIM, row0, 0,  lane);
  v16bf aIm1 = load_frag(xim, E_DIM, row0, 32, lane);
  int m = lane & 15, half = lane >> 4;
  for (int nt = 0; nt < 4; nt++) {
    v16bf bRe0 = load_frag(wre, E_DIM, nt * 16, 0,  lane);
    v16bf bRe1 = load_frag(wre, E_DIM, nt * 16, 32, lane);
    v16bf bIm0 = load_frag(wim, E_DIM, nt * 16, 0,  lane);
    v16bf bIm1 = load_frag(wim, E_DIM, nt * 16, 32, lane);
    v16bf bMn0 = load_frag(wmn, E_DIM, nt * 16, 0,  lane);
    v16bf bMn1 = load_frag(wmn, E_DIM, nt * 16, 32, lane);
    v8f accRe = {}; v8f accIm = {};
    accRe = WMMA_BF16(aRe0, bRe0, accRe); accRe = WMMA_BF16(aRe1, bRe1, accRe);
    accRe = WMMA_BF16(aIm0, bMn0, accRe); accRe = WMMA_BF16(aIm1, bMn1, accRe);
    accIm = WMMA_BF16(aRe0, bIm0, accIm); accIm = WMMA_BF16(aRe1, bIm1, accIm);
    accIm = WMMA_BF16(aIm0, bRe0, accIm); accIm = WMMA_BF16(aIm1, bRe1, accIm);
    for (int v = 0; v < 8; v++) {
      int bhr = row0 + v + 8 * half;
      int o = nt * 16 + m;
      size_t base = ((size_t)bhr * E_DIM + o) * (2 * NMODE);
      Yt[base + i]         = f2bf(accRe[v]);
      Yt[base + NMODE + i] = f2bf(accIm[v]);
    }
  }
}

// ---------------------------------------------------------------------------
// Kernel 3: sparse inverse DFT.  Per (bh, s-slab of 512) block:
//   out[s, e] = G[s, 0:128] @ Yt_bh[0:128, e]    (M=512/block, N=64, K=128)
// Streams the final 134 MB f32 output.
// ---------------------------------------------------------------------------
__global__ void __launch_bounds__(256)
fa_idft(const unsigned short* __restrict__ G, const unsigned short* __restrict__ Yt,
        float* __restrict__ out) {
  int bh = blockIdx.x;
  int s_block = blockIdx.y * 512;
  int lane = threadIdx.x & 31, wave = threadIdx.x >> 5;
  const unsigned short* yb = Yt + (size_t)bh * E_DIM * (2 * NMODE);
  float* ob = out + (size_t)bh * S_LEN * E_DIM;
  int m = lane & 15, half = lane >> 4;
  for (int nt = 0; nt < 4; nt++) {
    v16bf bfk[4];
    for (int kt = 0; kt < 4; kt++)
      bfk[kt] = load_frag(yb, 2 * NMODE, nt * 16, kt * 32, lane);
    for (int mi = 0; mi < 4; mi++) {
      int s0 = s_block + (wave * 4 + mi) * 16;
      v8f acc = {};
      for (int kt = 0; kt < 4; kt++) {
        v16bf a = load_frag(G, 2 * NMODE, s0, kt * 32, lane);
        acc = WMMA_BF16(a, bfk[kt], acc);
      }
      for (int v = 0; v < 8; v++) {
        int s = s0 + v + 8 * half;
        ob[(size_t)s * E_DIM + nt * 16 + m] = acc[v];
      }
    }
  }
}

// ---------------------------------------------------------------------------
extern "C" void kernel_launch(void* const* d_in, const int* in_sizes, int n_in,
                              void* d_out, int out_size, void* d_ws, size_t ws_size,
                              hipStream_t stream) {
  (void)in_sizes; (void)n_in; (void)out_size; (void)ws_size;
  const float* x  = (const float*)d_in[0];
  const float* wr = (const float*)d_in[1];
  const float* wi = (const float*)d_in[2];
  const int*   idx = (const int*)d_in[3];   // JAX default x64-off => int32
  float* out = (float*)d_out;

  // Workspace layout (bytes); total 7.5 MB, fully L2-resident.
  char* ws = (char*)d_ws;
  const size_t MB = 1u << 20, HK = 1u << 19;
  unsigned short* F2   = (unsigned short*)(ws + 0 * MB);          // 1 MB
  unsigned short* G    = (unsigned short*)(ws + 1 * MB);          // 1 MB
  unsigned short* Wre  = (unsigned short*)(ws + 2 * MB);          // 512 KB
  unsigned short* Wim  = (unsigned short*)(ws + 2 * MB + 1 * HK); // 512 KB
  unsigned short* WimN = (unsigned short*)(ws + 2 * MB + 2 * HK); // 512 KB
  unsigned short* Xre  = (unsigned short*)(ws + 2 * MB + 3 * HK); // 1 MB
  unsigned short* Xim  = (unsigned short*)(ws + 2 * MB + 5 * HK); // 1 MB
  unsigned short* Yt   = (unsigned short*)(ws + 2 * MB + 7 * HK); // 2 MB

  // 0: twiddles + weight transpose (1,310,720 work items)
  fa_precompute<<<5120, 256, 0, stream>>>(wr, wi, idx, F2, G, Wre, Wim, WimN);
  // 1: forward sparse DFT (reads x, 134 MB)
  fa_dft<<<dim3(BH, 2), 256, 0, stream>>>(x, F2, Xre, Xim);
  // 2: per-mode complex mix
  fa_mix<<<NMODE, 256, 0, stream>>>(Xre, Xim, Wre, Wim, WimN, Yt);
  // 3: inverse sparse DFT (writes out, 134 MB)
  fa_idft<<<dim3(BH, 8), 256, 0, stream>>>(G, Yt, out);
}
